// Router_16621523435664
// MI455X (gfx1250) — matrix operand order repaired and checked by reading, not compiled, not run on hardware
//
#include <hip/hip_runtime.h>
#include <hip/hip_bf16.h>

typedef __attribute__((ext_vector_type(16))) __bf16 v16bf;
typedef __attribute__((ext_vector_type(8)))  __bf16 v8bf;
typedef __attribute__((ext_vector_type(8)))  float  v8f;

__device__ __forceinline__ __bf16 f2bf(float f) {
  union { float f; unsigned u; } a; a.f = f;
  unsigned r = a.u + 0x7FFFu + ((a.u >> 16) & 1u);   // round-to-nearest-even
  union { unsigned short s; __bf16 b; } o; o.s = (unsigned short)(r >> 16);
  return o.b;
}

// ---------- CDNA5 async global->LDS copy (ASYNCcnt domain, bypasses VGPRs) ----------
__device__ __forceinline__ void async_b128(const __bf16* g, const __bf16* l) {
  // low 32 bits of a generic LDS pointer == LDS byte offset (shared aperture layout)
  unsigned lofs = (unsigned)(uintptr_t)l;
  asm volatile("global_load_async_to_lds_b128 %0, %1, off"
               :: "v"(lofs), "v"(g)
               : "memory");
}

__device__ __forceinline__ void wait_async0() {
#if __has_builtin(__builtin_amdgcn_s_wait_asynccnt)
  __builtin_amdgcn_s_wait_asynccnt(0);
#else
  asm volatile("s_wait_asynccnt 0x0" ::: "memory");
#endif
}

// ---------------- elementwise fp32 -> bf16 ----------------
__global__ void k_convert_bf16(const float* __restrict__ in, __bf16* __restrict__ out, int n) {
  int i = blockIdx.x * 256 + threadIdx.x;
  if (i < n) out[i] = f2bf(in[i]);
}

// ------------- tiled transpose: fp32 [R][C] -> bf16 [C][R] -------------
__global__ void __launch_bounds__(256)
k_transpose_bf16(const float* __restrict__ in, __bf16* __restrict__ out, int R, int C) {
  __shared__ __bf16 t[64][68];
  const int c0 = blockIdx.x * 64, r0 = blockIdx.y * 64;
  const int tid = threadIdx.x;
  const int cl = tid & 63, rw = tid >> 6;
#pragma unroll
  for (int it = 0; it < 16; ++it) {
    int r = rw + it * 4;
    t[r][cl] = f2bf(in[(size_t)(r0 + r) * C + c0 + cl]);
  }
  __syncthreads();
  const int rl = tid & 63, cw = tid >> 6;
#pragma unroll
  for (int it = 0; it < 16; ++it) {
    int c = cw + it * 4;
    out[(size_t)(c0 + c) * R + r0 + rl] = t[rl][c];
  }
}

// ---------------- router: p = sigmoid(x @ Wr + br), wave-per-row ----------------
__global__ void __launch_bounds__(256)
k_router(const float* __restrict__ x, const float* __restrict__ Wr,
         const float* __restrict__ br, float* __restrict__ P, int D) {
  const int wave = threadIdx.x >> 5, lane = threadIdx.x & 31;
  const int row = blockIdx.x * 8 + wave;
  const float* xr = x + (size_t)row * D;
  float s = 0.f;
  for (int i = lane; i < D; i += 32) s += xr[i] * Wr[i];
#pragma unroll
  for (int off = 16; off > 0; off >>= 1) s += __shfl_xor(s, off, 32);
  if (lane == 0) P[row] = 1.f / (1.f + __expf(-(s + br[0])));
}

// ---------------- bf16 WMMA GEMM core ----------------
// A: [M,K] bf16 row-major; BT: [N,K] bf16 row-major (pre-transposed weights).
// 256 thr = 8 wave32; block tile 128x128; per-wave 32x64 = 2x4 WMMA tiles.
// Double-buffered LDS stages of K=64, filled by async direct-to-LDS copies:
//   per stage: 1x s_wait_asynccnt + 1 barrier, 16 WMMAs + 24 ds_load_b128 per wave.
#define BM 128
#define BN 128
#define SK 64      // K elements per stage (2 WMMA k-slices)
#define SKP 72     // padded LDS row stride (16B-aligned, spreads banks)
#define BUFE (BM * SKP)   // elements per LDS buffer

__device__ __forceinline__ void wmma_pipeline(
    const __bf16* __restrict__ A, const __bf16* __restrict__ BT,
    int m0, int n0, int K, int tid,
    __bf16* As, __bf16* Bs, v8f acc[2][4]) {
  const int lane  = tid & 31;
  const int wave  = tid >> 5;
  const int waveM = wave >> 1;   // 0..3
  const int waveN = wave & 1;    // 0..1
  const int half  = lane >> 4;
  const int l15   = lane & 15;

  // per-thread staging chunks: 128x64 bf16 tile = 1024 b128 chunks; 4 per thread
  int rowC[4], koC[4];
#pragma unroll
  for (int i = 0; i < 4; ++i) {
    int c = tid + i * 256;
    rowC[i] = c >> 3;           // 0..127
    koC[i]  = (c & 7) * 8;      // 0..56, 8-element (16B) granules
  }

  const int nst = K / SK;

  // prologue: stage 0 -> buffer 0
#pragma unroll
  for (int i = 0; i < 4; ++i) {
    async_b128(&A [(size_t)(m0 + rowC[i]) * K + koC[i]], &As[rowC[i] * SKP + koC[i]]);
    async_b128(&BT[(size_t)(n0 + rowC[i]) * K + koC[i]], &Bs[rowC[i] * SKP + koC[i]]);
  }

  for (int s = 0; s < nst; ++s) {
    wait_async0();        // my stage-s tiles landed in LDS
    __syncthreads();      // everyone's landed; previous readers of next buffer done
    const int buf = s & 1;
    if (s + 1 < nst) {    // kick off stage s+1 into the other buffer (overlaps WMMAs)
      const int k0 = (s + 1) * SK;
      const int nb = (buf ^ 1) * BUFE;
#pragma unroll
      for (int i = 0; i < 4; ++i) {
        async_b128(&A [(size_t)(m0 + rowC[i]) * K + k0 + koC[i]], &As[nb + rowC[i] * SKP + koC[i]]);
        async_b128(&BT[(size_t)(n0 + rowC[i]) * K + k0 + koC[i]], &Bs[nb + rowC[i] * SKP + koC[i]]);
      }
    }
    const __bf16* Ab = As + buf * BUFE;
    const __bf16* Bb = Bs + buf * BUFE;
#pragma unroll
    for (int ks = 0; ks < SK; ks += 32) {
      v16bf aF[2], bF[4];
#pragma unroll
      for (int i = 0; i < 2; ++i) {
        const __bf16* base = Ab + (waveM * 32 + i * 16 + l15) * SKP + ks;
        v8bf lo = *(const v8bf*)(base + half * 8);        // K = 8h .. 8h+7
        v8bf hi = *(const v8bf*)(base + 16 + half * 8);   // K = 8h+16 .. 8h+23
        aF[i] = __builtin_shufflevector(lo, hi, 0,1,2,3,4,5,6,7,8,9,10,11,12,13,14,15);
      }
#pragma unroll
      for (int j = 0; j < 4; ++j) {
        const __bf16* base = Bb + (waveN * 64 + j * 16 + l15) * SKP + ks;
        v8bf lo = *(const v8bf*)(base + half * 8);        // K = 8h .. 8h+7
        v8bf hi = *(const v8bf*)(base + 16 + half * 8);   // K = 8h+16 .. 8h+23
        bF[j] = __builtin_shufflevector(lo, hi, 0,1,2,3,4,5,6,7,8,9,10,11,12,13,14,15);
      }
#pragma unroll
      for (int i = 0; i < 2; ++i)
#pragma unroll
        for (int j = 0; j < 4; ++j)
          acc[i][j] = __builtin_amdgcn_wmma_f32_16x16x32_bf16(
              false, aF[i], false, bF[j], (short)0, acc[i][j], false, false);
    }
  }
}

// GEMM1: h = relu(A @ B + bias) -> bf16
__global__ void __launch_bounds__(256)
k_gemm_relu(const __bf16* __restrict__ A, const __bf16* __restrict__ BT,
            const float* __restrict__ bias, __bf16* __restrict__ Hout,
            int N, int K) {
  __shared__ alignas(128) __bf16 As[2 * BUFE];
  __shared__ alignas(128) __bf16 Bs[2 * BUFE];
  const int tid = threadIdx.x;
  const int m0 = blockIdx.y * BM, n0 = blockIdx.x * BN;
  const int lane = tid & 31, wave = tid >> 5;
  const int waveM = wave >> 1, waveN = wave & 1;
  const int half = lane >> 4, l15 = lane & 15;

  v8f acc[2][4];
  const v8f vz = {0.f, 0.f, 0.f, 0.f, 0.f, 0.f, 0.f, 0.f};
#pragma unroll
  for (int i = 0; i < 2; ++i)
#pragma unroll
    for (int j = 0; j < 4; ++j) acc[i][j] = vz;

  wmma_pipeline(A, BT, m0, n0, K, tid, As, Bs, acc);

#pragma unroll
  for (int j = 0; j < 4; ++j) {
    int col = n0 + waveN * 64 + j * 16 + l15;
    float bv = bias[col];
#pragma unroll
    for (int i = 0; i < 2; ++i) {
      int rbase = m0 + waveM * 32 + i * 16 + half * 8;
#pragma unroll
      for (int r = 0; r < 8; ++r) {
        float v = acc[i][j][r] + bv;
        v = v > 0.f ? v : 0.f;
        Hout[(size_t)(rbase + r) * N + col] = f2bf(v);
      }
    }
  }
}

// GEMM2: mode 0: out = p*(A@B + bias); mode 1: out += (1-p)*(A@B + bias)
__global__ void __launch_bounds__(256)
k_gemm_route(const __bf16* __restrict__ A, const __bf16* __restrict__ BT,
             const float* __restrict__ bias, const float* __restrict__ P,
             float* __restrict__ out, int N, int K, int mode) {
  __shared__ alignas(128) __bf16 As[2 * BUFE];
  __shared__ alignas(128) __bf16 Bs[2 * BUFE];
  const int tid = threadIdx.x;
  const int m0 = blockIdx.y * BM, n0 = blockIdx.x * BN;
  const int lane = tid & 31, wave = tid >> 5;
  const int waveM = wave >> 1, waveN = wave & 1;
  const int half = lane >> 4, l15 = lane & 15;

  v8f acc[2][4];
  const v8f vz = {0.f, 0.f, 0.f, 0.f, 0.f, 0.f, 0.f, 0.f};
#pragma unroll
  for (int i = 0; i < 2; ++i)
#pragma unroll
    for (int j = 0; j < 4; ++j) acc[i][j] = vz;

  wmma_pipeline(A, BT, m0, n0, K, tid, As, Bs, acc);

#pragma unroll
  for (int j = 0; j < 4; ++j) {
    int col = n0 + waveN * 64 + j * 16 + l15;
    float bv = bias[col];
#pragma unroll
    for (int i = 0; i < 2; ++i) {
      int rbase = m0 + waveM * 32 + i * 16 + half * 8;
#pragma unroll
      for (int r = 0; r < 8; ++r) {
        int row = rbase + r;
        float pv = P[row];
        float sc = mode ? (1.f - pv) : pv;
        float v  = sc * (acc[i][j][r] + bv);
        size_t idx = (size_t)row * N + col;
        if (mode) out[idx] += v; else out[idx] = v;
      }
    }
  }
}

// ---------------- host-side orchestration ----------------
extern "C" void kernel_launch(void* const* d_in, const int* in_sizes, int n_in,
                              void* d_out, int out_size, void* d_ws, size_t ws_size,
                              hipStream_t stream) {
  const int Bn = 8192, D = 1024, H = 4096;
  const float* x   = (const float*)d_in[0];
  const float* Wr  = (const float*)d_in[1];
  const float* br  = (const float*)d_in[2];
  const float* W1l = (const float*)d_in[3];
  const float* b1l = (const float*)d_in[4];
  const float* W2l = (const float*)d_in[5];
  const float* b2l = (const float*)d_in[6];
  const float* W1r = (const float*)d_in[7];
  const float* b1r = (const float*)d_in[8];
  const float* W2r = (const float*)d_in[9];
  const float* b2r = (const float*)d_in[10];
  float* out = (float*)d_out;

  // workspace: P (32KB) | xb 16MB | hb 64MB | w1t 8MB | w2t 8MB  (~96MB total)
  char* ws = (char*)d_ws;
  float*  P   = (float*)ws;
  __bf16* xb  = (__bf16*)(ws + 32768);
  __bf16* hb  = (__bf16*)(ws + 32768 + (size_t)Bn * D * 2);
  __bf16* w1t = (__bf16*)(ws + 32768 + (size_t)Bn * D * 2 + (size_t)Bn * H * 2);
  __bf16* w2t = w1t + (size_t)H * D;

  dim3 blk(256);
  k_convert_bf16<<<dim3((Bn * D + 255) / 256), blk, 0, stream>>>(x, xb, Bn * D);
  k_router<<<dim3(Bn / 8), blk, 0, stream>>>(x, Wr, br, P, D);

  dim3 g_t1(H / 64, D / 64);   // W1 [D,H] -> [H,D]
  dim3 g_t2(D / 64, H / 64);   // W2 [H,D] -> [D,H]
  dim3 g_g1(H / BN, Bn / BM);  // GEMM1: N=H
  dim3 g_g2(D / BN, Bn / BM);  // GEMM2: N=D

  // ---- left expert ----
  k_transpose_bf16<<<g_t1, blk, 0, stream>>>(W1l, w1t, D, H);
  k_transpose_bf16<<<g_t2, blk, 0, stream>>>(W2l, w2t, H, D);
  k_gemm_relu <<<g_g1, blk, 0, stream>>>(xb, w1t, b1l, hb, H, D);
  k_gemm_route<<<g_g2, blk, 0, stream>>>(hb, w2t, b2l, P, out, D, H, 0);

  // ---- right expert (reuses hb / w1t / w2t; stream order serializes) ----
  k_transpose_bf16<<<g_t1, blk, 0, stream>>>(W1r, w1t, D, H);
  k_transpose_bf16<<<g_t2, blk, 0, stream>>>(W2r, w2t, H, D);
  k_gemm_relu <<<g_g1, blk, 0, stream>>>(xb, w1t, b1r, hb, H, D);
  k_gemm_route<<<g_g2, blk, 0, stream>>>(hb, w2t, b2r, P, out, D, H, 1);
}
